// TransformerDecoderLayer_49091476194114
// MI455X (gfx1250) — matrix operand order repaired and checked by reading, not compiled, not run on hardware
//
#include <hip/hip_runtime.h>
#include <hip/hip_bf16.h>

typedef _Float16 f16;
typedef _Float16 v16h __attribute__((ext_vector_type(16)));
typedef float    v8f  __attribute__((ext_vector_type(8)));

__device__ __forceinline__ v8f v8f_zero() {
  v8f z;
#pragma unroll
  for (int i = 0; i < 8; ++i) z[i] = 0.f;
  return z;
}

__device__ __forceinline__ v8f wmma16(v16h a, v16h b, v8f c) {
  // D = A(16x32 f16) * B(32x16 f16) + C(16x16 f32)
  return __builtin_amdgcn_wmma_f32_16x16x32_f16(false, a, false, b, (short)0, c,
                                                false, false);
}

// Async 16B global -> LDS copy (per lane), tracked by ASYNCcnt.
// LDS address = low 32 bits of the generic pointer (flat->LDS aperture rule).
__device__ __forceinline__ void async_copy_b128(const f16* gptr, f16* lptr) {
  const unsigned laddr = (unsigned)(size_t)lptr;
  asm volatile("global_load_async_to_lds_b128 %0, %1, off"
               :
               : "v"(laddr), "v"(gptr)
               : "memory");
}
__device__ __forceinline__ void async_wait0() {
  asm volatile("s_wait_asynccnt 0x0" ::: "memory");
}

// A fragment (16x32, MxK), row-major source with leading dim ld (halves).
// Lane l: M = l&15, K-halves: l<16 -> {0..7,16..23}, l>=16 -> {8..15,24..31}.
// Each run of 8 K-halves is contiguous -> two 16B loads.
__device__ __forceinline__ v16h load_a_frag(const f16* p, int ld, int lane) {
  const int m = lane & 15;
  const int g = (lane >> 4) << 3;
  const f16* rp = p + m * ld + g;
  union { uint4 u[2]; v16h v; } t;
  t.u[0] = *(const uint4*)(rp);
  t.u[1] = *(const uint4*)(rp + 16);
  return t.v;
}

// B fragment where logical B[k][n] = p[n*ld + k]  (source stored [N][K] row
// major, i.e. K-contiguous). Lane l: N = l&15, K-halves: l<16 -> {0..7,16..23},
// l>=16 -> {8..15,24..31} -> two 16B
// loads. Requires ld multiple of 8 halves for 16B alignment.
__device__ __forceinline__ v16h load_bT_frag(const f16* p, int ld, int lane) {
  const int n  = lane & 15;
  const int kb = (lane >> 4) << 3;
  const f16* rp = p + n * ld + kb;
  union { uint4 u[2]; v16h v; } t;
  t.u[0] = *(const uint4*)(rp);
  t.u[1] = *(const uint4*)(rp + 16);
  return t.v;
}

// ---------------------------------------------------------------- cvt ------
__global__ void cvt_kernel(const float* __restrict__ in, f16* __restrict__ out,
                           long n) {
  long i = (long)blockIdx.x * blockDim.x + threadIdx.x;
  const long stride = (long)gridDim.x * blockDim.x;
  for (; i < n; i += stride) out[i] = (f16)in[i];
}

// ---------------------------------------------------------------- GEMM -----
// C[M,N] = act(A[M,K] @ W[K,N] + bias [+ residual]);  A,W f16 row-major.
// 256 threads = 8 waves; block tile 128x128; wave tile 64x32; K-step 32.
// A tile: async DMA to LDS.  W tile: staged transposed (K-contiguous) so all
// fragment loads are ds_load_b128.
__global__ __launch_bounds__(256) void gemm_f16_kernel(
    const f16* __restrict__ A, const f16* __restrict__ W,
    const float* __restrict__ bias, const float* __restrict__ resid,
    float* __restrict__ outF, f16* __restrict__ outH, int M, int N, int K,
    int relu) {
  __shared__ f16 As[128][32];
  __shared__ f16 BsT[128][40];   // [n][k], padded to 40 halves (80B rows)
  const int tid = threadIdx.x;
  const int lane = tid & 31, w = tid >> 5;
  const int wr = w >> 2, wc = w & 3;        // 2x4 wave grid
  const int bm = blockIdx.y * 128, bn = blockIdx.x * 128;

  // per-thread staging units (straight-line, no dynamic loop)
  const int ar = tid >> 2, ac = (tid & 3) << 3;    // A: rows ar, ar+64
  const int br = tid >> 4, bc = (tid & 15) << 3;   // W: k rows br, br+16

  v8f acc[4][2];
#pragma unroll
  for (int i = 0; i < 4; ++i)
#pragma unroll
    for (int j = 0; j < 2; ++j) acc[i][j] = v8f_zero();

  for (int k0 = 0; k0 < K; k0 += 32) {
    // A tile: async global->LDS (no VGPR round trip)
    async_copy_b128(&A[(long)(bm + ar) * K + k0 + ac], &As[ar][ac]);
    async_copy_b128(&A[(long)(bm + ar + 64) * K + k0 + ac], &As[ar + 64][ac]);

    // W tile: load 2x16B, transpose-scatter into BsT
    const uint4 t0 = *(const uint4*)&W[(long)(k0 + br) * N + bn + bc];
    const uint4 t1 = *(const uint4*)&W[(long)(k0 + br + 16) * N + bn + bc];
    union { uint4 q; f16 h[8]; } u0, u1;
    u0.q = t0; u1.q = t1;
#pragma unroll
    for (int i = 0; i < 8; ++i) BsT[bc + i][br] = u0.h[i];
#pragma unroll
    for (int i = 0; i < 8; ++i) BsT[bc + i][br + 16] = u1.h[i];

    // prefetch next K-step while this tile computes
    const int kn = k0 + 32;
    if (kn < K) {
      __builtin_prefetch(&A[(long)(bm + ar) * K + kn + ac], 0, 0);
      __builtin_prefetch(&A[(long)(bm + ar + 64) * K + kn + ac], 0, 0);
      __builtin_prefetch(&W[(long)(kn + br) * N + bn + bc], 0, 0);
      __builtin_prefetch(&W[(long)(kn + br + 16) * N + bn + bc], 0, 0);
    }
    async_wait0();
    __syncthreads();

    v16h af[4], bf[2];
#pragma unroll
    for (int mt = 0; mt < 4; ++mt)
      af[mt] = load_a_frag(&As[wr * 64 + mt * 16][0], 32, lane);
#pragma unroll
    for (int nt = 0; nt < 2; ++nt)
      bf[nt] = load_bT_frag(&BsT[wc * 32 + nt * 16][0], 40, lane);
#pragma unroll
    for (int mt = 0; mt < 4; ++mt)
#pragma unroll
      for (int nt = 0; nt < 2; ++nt)
        acc[mt][nt] = wmma16(af[mt], bf[nt], acc[mt][nt]);
    __syncthreads();
  }

  const int hi = lane >> 4, ln = lane & 15;
#pragma unroll
  for (int mt = 0; mt < 4; ++mt)
#pragma unroll
    for (int nt = 0; nt < 2; ++nt)
#pragma unroll
      for (int e = 0; e < 8; ++e) {
        const int gm = bm + wr * 64 + mt * 16 + e + hi * 8;
        const int gn = bn + wc * 32 + nt * 16 + ln;
        float v = acc[mt][nt][e] + bias[gn];
        if (resid) v += resid[(long)gm * N + gn];
        if (relu) v = v > 0.f ? v : 0.f;
        if (outF) outF[(long)gm * N + gn] = v;
        if (outH) outH[(long)gm * N + gn] = (f16)v;
      }
}

// ------------------------------------------------------------ attention ----
// Flash-attention. Grid (L/64, H, B), 128 threads (4 waves).
// Wave w handles 16 query rows; K-blocks of 32 staged in LDS.
// K tile: async DMA to LDS (K-contiguous for Q.K^T). V tile staged transposed
// (K-contiguous for P.V) so all fragment loads are ds_load_b128.
__global__ __launch_bounds__(128) void attention_kernel(
    const f16* __restrict__ Q, const f16* __restrict__ Kh,
    const f16* __restrict__ Vh, f16* __restrict__ O, int L, int causal) {
  __shared__ f16 Ks[32][64];     // [k][d]
  __shared__ f16 VsT[64][40];    // [d][k], padded
  __shared__ f16 Ps[4][16][32];  // per-wave P tile
  const int tid = threadIdx.x, lane = tid & 31, w = tid >> 5;
  const int hi = lane >> 4, ln = lane & 15;
  const int hc = blockIdx.y * 64;                 // head column offset
  const long brow = (long)blockIdx.z * L;         // batch row offset
  const int q0 = blockIdx.x * 64 + w * 16;        // this wave's first q row
  const float scale = 0.125f;                     // 1/sqrt(64)
  const int kr = tid >> 3, kc = (tid & 7) << 3;   // staging units: kr, kr+16

  v16h qf[2];
#pragma unroll
  for (int c = 0; c < 2; ++c)
    qf[c] = load_a_frag(Q + (brow + q0) * 1024 + hc + c * 32, 1024, lane);

  v8f o[4];
#pragma unroll
  for (int i = 0; i < 4; ++i) o[i] = v8f_zero();
  float ms[8], ls[8];
#pragma unroll
  for (int r = 0; r < 8; ++r) { ms[r] = -3e38f; ls[r] = 0.f; }

  const int kmax = causal ? (blockIdx.x * 64 + 64) : L;
  for (int k0 = 0; k0 < kmax; k0 += 32) {
    const long g0 = (brow + k0 + kr) * 1024 + hc + kc;
    const long g1 = (brow + k0 + kr + 16) * 1024 + hc + kc;
    async_copy_b128(&Kh[g0], &Ks[kr][kc]);
    async_copy_b128(&Kh[g1], &Ks[kr + 16][kc]);
    union { uint4 q; f16 h[8]; } v0u, v1u;
    v0u.q = *(const uint4*)&Vh[g0];
    v1u.q = *(const uint4*)&Vh[g1];
#pragma unroll
    for (int i = 0; i < 8; ++i) VsT[kc + i][kr] = v0u.h[i];
#pragma unroll
    for (int i = 0; i < 8; ++i) VsT[kc + i][kr + 16] = v1u.h[i];
    if (k0 + 32 < kmax) {
      __builtin_prefetch(&Kh[g0 + 32 * 1024], 0, 0);
      __builtin_prefetch(&Vh[g0 + 32 * 1024], 0, 0);
      __builtin_prefetch(&Kh[g1 + 32 * 1024], 0, 0);
      __builtin_prefetch(&Vh[g1 + 32 * 1024], 0, 0);
    }
    async_wait0();
    __syncthreads();

    // S (16q x 32k) = Q (16x64) . K^T (64x32)
    v8f s[2];
    s[0] = v8f_zero(); s[1] = v8f_zero();
#pragma unroll
    for (int nt = 0; nt < 2; ++nt)
#pragma unroll
      for (int c = 0; c < 2; ++c)
        s[nt] = wmma16(qf[c], load_bT_frag(&Ks[nt * 16][c * 32], 64, lane),
                       s[nt]);

    // online softmax per row (rows of this lane: r + 8*hi; cols = ln, 16+ln)
    float alpha[8];
#pragma unroll
    for (int r = 0; r < 8; ++r) {
      float v0 = s[0][r] * scale;
      float v1 = s[1][r] * scale;
      if (causal) {
        const int gq = q0 + r + hi * 8;
        if (k0 + ln > gq) v0 = -1e9f;
        if (k0 + 16 + ln > gq) v1 = -1e9f;
      }
      float rm = fmaxf(v0, v1);
      rm = fmaxf(rm, __shfl_xor(rm, 1, 32));
      rm = fmaxf(rm, __shfl_xor(rm, 2, 32));
      rm = fmaxf(rm, __shfl_xor(rm, 4, 32));
      rm = fmaxf(rm, __shfl_xor(rm, 8, 32));
      const float mnew = fmaxf(ms[r], rm);
      const float a = __expf(ms[r] - mnew);
      const float p0 = __expf(v0 - mnew);
      const float p1 = __expf(v1 - mnew);
      float rs = p0 + p1;
      rs += __shfl_xor(rs, 1, 32);
      rs += __shfl_xor(rs, 2, 32);
      rs += __shfl_xor(rs, 4, 32);
      rs += __shfl_xor(rs, 8, 32);
      ls[r] = ls[r] * a + rs;
      ms[r] = mnew;
      alpha[r] = a;
      Ps[w][r + hi * 8][ln] = (f16)p0;            // C-layout -> A-layout via LDS
      Ps[w][r + hi * 8][16 + ln] = (f16)p1;
    }
#pragma unroll
    for (int on = 0; on < 4; ++on)
#pragma unroll
      for (int r = 0; r < 8; ++r) o[on][r] *= alpha[r];

    // O (16x64) += P (16x32) . V (32x64)
    const v16h pa = load_a_frag(&Ps[w][0][0], 32, lane);
#pragma unroll
    for (int on = 0; on < 4; ++on)
      o[on] = wmma16(pa, load_bT_frag(&VsT[on * 16][0], 40, lane), o[on]);
    __syncthreads();
  }

#pragma unroll
  for (int r = 0; r < 8; ++r) ls[r] = 1.f / ls[r];
#pragma unroll
  for (int on = 0; on < 4; ++on)
#pragma unroll
    for (int r = 0; r < 8; ++r) {
      const long gm = brow + q0 + r + hi * 8;
      const int gn = hc + on * 16 + ln;
      O[gm * 1024 + gn] = (f16)(o[on][r] * ls[r]);
    }
}

// ------------------------------------------------------------ layernorm ----
__global__ __launch_bounds__(256) void layernorm_kernel(
    const float* __restrict__ in, const float* __restrict__ g,
    const float* __restrict__ b, float* __restrict__ outF,
    f16* __restrict__ outH) {
  __shared__ float ws0[8], ws1[8];
  const long base = (long)blockIdx.x * 1024;
  const int w = threadIdx.x >> 5, lane = threadIdx.x & 31;
  float x[4], s = 0.f, s2 = 0.f;
#pragma unroll
  for (int i = 0; i < 4; ++i) {
    x[i] = in[base + threadIdx.x + 256 * i];
    s += x[i];
    s2 += x[i] * x[i];
  }
#pragma unroll
  for (int m = 1; m < 32; m <<= 1) {
    s += __shfl_xor(s, m, 32);
    s2 += __shfl_xor(s2, m, 32);
  }
  if (lane == 0) { ws0[w] = s; ws1[w] = s2; }
  __syncthreads();
  if (w == 0) {
    float a = lane < 8 ? ws0[lane] : 0.f;
    float c = lane < 8 ? ws1[lane] : 0.f;
#pragma unroll
    for (int m = 1; m < 8; m <<= 1) {
      a += __shfl_xor(a, m, 32);
      c += __shfl_xor(c, m, 32);
    }
    if (lane == 0) { ws0[0] = a; ws1[0] = c; }
  }
  __syncthreads();
  const float mu = ws0[0] * (1.f / 1024.f);
  const float var = ws1[0] * (1.f / 1024.f) - mu * mu;
  const float rinv = rsqrtf(var + 1e-5f);
#pragma unroll
  for (int i = 0; i < 4; ++i) {
    const int cidx = threadIdx.x + 256 * i;
    const float v = (x[i] - mu) * rinv * g[cidx] + b[cidx];
    if (outF) outF[base + cidx] = v;
    if (outH) outH[base + cidx] = (f16)v;
  }
}

// ---------------------------------------------------------------- driver ---
extern "C" void kernel_launch(void* const* d_in, const int* in_sizes, int n_in,
                              void* d_out, int out_size, void* d_ws,
                              size_t ws_size, hipStream_t stream) {
  (void)in_sizes; (void)n_in; (void)out_size; (void)ws_size;
  const float* tgt = (const float*)d_in[0];
  const float* mem = (const float*)d_in[1];
  // d_in[2] = tgt_mask (causal; handled analytically)
  const float* sa_wq = (const float*)d_in[3];
  const float* sa_bq = (const float*)d_in[4];
  const float* sa_wk = (const float*)d_in[5];
  const float* sa_bk = (const float*)d_in[6];
  const float* sa_wv = (const float*)d_in[7];
  const float* sa_bv = (const float*)d_in[8];
  const float* sa_wo = (const float*)d_in[9];
  const float* sa_bo = (const float*)d_in[10];
  const float* ca_wq = (const float*)d_in[11];
  const float* ca_bq = (const float*)d_in[12];
  const float* ca_wk = (const float*)d_in[13];
  const float* ca_bk = (const float*)d_in[14];
  const float* ca_wv = (const float*)d_in[15];
  const float* ca_bv = (const float*)d_in[16];
  const float* ca_wo = (const float*)d_in[17];
  const float* ca_bo = (const float*)d_in[18];
  const float* ffn_w1 = (const float*)d_in[19];
  const float* ffn_b1 = (const float*)d_in[20];
  const float* ffn_w2 = (const float*)d_in[21];
  const float* ffn_b2 = (const float*)d_in[22];
  const float* ln1_g = (const float*)d_in[23];
  const float* ln1_b = (const float*)d_in[24];
  const float* ln2_g = (const float*)d_in[25];
  const float* ln2_b = (const float*)d_in[26];
  const float* ln3_g = (const float*)d_in[27];
  const float* ln3_b = (const float*)d_in[28];

  const long M = 4096;  // B*L
  char* ws = (char*)d_ws;
  size_t off = 0;
  auto alloc = [&](size_t bytes) -> void* {
    void* p = ws + off;
    off += (bytes + 255) & ~(size_t)255;
    return p;
  };
  f16* tgt_h = (f16*)alloc(M * 1024 * 2);
  f16* mem_h = (f16*)alloc(M * 1024 * 2);
  f16* wqh = (f16*)alloc(1024 * 1024 * 2);
  f16* wkh = (f16*)alloc(1024 * 1024 * 2);
  f16* wvh = (f16*)alloc(1024 * 1024 * 2);
  f16* woh = (f16*)alloc(1024 * 1024 * 2);
  f16* cwqh = (f16*)alloc(1024 * 1024 * 2);
  f16* cwkh = (f16*)alloc(1024 * 1024 * 2);
  f16* cwvh = (f16*)alloc(1024 * 1024 * 2);
  f16* cwoh = (f16*)alloc(1024 * 1024 * 2);
  f16* w1h = (f16*)alloc(1024L * 4096 * 2);
  f16* w2h = (f16*)alloc(4096L * 1024 * 2);
  f16* q_h = (f16*)alloc(M * 1024 * 2);
  f16* k_h = (f16*)alloc(M * 1024 * 2);
  f16* v_h = (f16*)alloc(M * 1024 * 2);
  f16* ao_h = (f16*)alloc(M * 1024 * 2);
  f16* x_h = (f16*)alloc(M * 1024 * 2);
  f16* hid_h = (f16*)alloc(M * 4096 * 2);
  float* res_f = (float*)alloc(M * 1024 * 4);
  float* x_f = (float*)alloc(M * 1024 * 4);

  auto cvt = [&](const float* src, f16* dst, long n) {
    int blocks = (int)((n + 2047) / 2048);
    cvt_kernel<<<blocks, 256, 0, stream>>>(src, dst, n);
  };
  auto gemm = [&](const f16* A, const f16* W, const float* bias,
                  const float* resid, float* oF, f16* oH, int m, int n, int k,
                  int relu) {
    dim3 grid(n / 128, m / 128);
    gemm_f16_kernel<<<grid, 256, 0, stream>>>(A, W, bias, resid, oF, oH, m, n,
                                              k, relu);
  };

  // precision conversion of activations + weights (once)
  cvt(tgt, tgt_h, M * 1024);
  cvt(mem, mem_h, M * 1024);
  cvt(sa_wq, wqh, 1024 * 1024);
  cvt(sa_wk, wkh, 1024 * 1024);
  cvt(sa_wv, wvh, 1024 * 1024);
  cvt(sa_wo, woh, 1024 * 1024);
  cvt(ca_wq, cwqh, 1024 * 1024);
  cvt(ca_wk, cwkh, 1024 * 1024);
  cvt(ca_wv, cwvh, 1024 * 1024);
  cvt(ca_wo, cwoh, 1024 * 1024);
  cvt(ffn_w1, w1h, 1024L * 4096);
  cvt(ffn_w2, w2h, 4096L * 1024);

  // ---- self attention ----
  gemm(tgt_h, wqh, sa_bq, nullptr, nullptr, q_h, 4096, 1024, 1024, 0);
  gemm(tgt_h, wkh, sa_bk, nullptr, nullptr, k_h, 4096, 1024, 1024, 0);
  gemm(tgt_h, wvh, sa_bv, nullptr, nullptr, v_h, 4096, 1024, 1024, 0);
  attention_kernel<<<dim3(16, 16, 4), 128, 0, stream>>>(q_h, k_h, v_h, ao_h,
                                                        1024, 1);
  gemm(ao_h, woh, sa_bo, tgt, res_f, nullptr, 4096, 1024, 1024, 0);
  layernorm_kernel<<<4096, 256, 0, stream>>>(res_f, ln1_g, ln1_b, x_f, x_h);

  // ---- cross attention ----
  gemm(x_h, cwqh, ca_bq, nullptr, nullptr, q_h, 4096, 1024, 1024, 0);
  gemm(mem_h, cwkh, ca_bk, nullptr, nullptr, k_h, 4096, 1024, 1024, 0);
  gemm(mem_h, cwvh, ca_bv, nullptr, nullptr, v_h, 4096, 1024, 1024, 0);
  attention_kernel<<<dim3(16, 16, 4), 128, 0, stream>>>(q_h, k_h, v_h, ao_h,
                                                        1024, 0);
  gemm(ao_h, cwoh, ca_bo, x_f, res_f, nullptr, 4096, 1024, 1024, 0);
  layernorm_kernel<<<4096, 256, 0, stream>>>(res_f, ln2_g, ln2_b, x_f, x_h);

  // ---- FFN ----
  gemm(x_h, w1h, ffn_b1, nullptr, nullptr, hid_h, 4096, 4096, 1024, 1);
  gemm(hid_h, w2h, ffn_b2, x_f, res_f, nullptr, 4096, 1024, 4096, 0);
  layernorm_kernel<<<4096, 256, 0, stream>>>(res_f, ln3_g, ln3_b,
                                             (float*)d_out, nullptr);
}
